// Message_Passer_NNM_1099511627808
// MI455X (gfx1250) — matrix operand order repaired and checked by reading, not compiled, not run on hardware
//
#include <hip/hip_runtime.h>

typedef __attribute__((ext_vector_type(16))) __bf16 v16bf;
typedef __attribute__((ext_vector_type(2)))  __bf16 v2bf;
typedef __attribute__((ext_vector_type(8)))  float  v8f;

#define NODE_DIM 32
#define EDGE_DIM 64
#define NN 1024                          // flat output cols of W (32*32)
#define N_TILES_TOTAL 4096               // 65536 edges / 16 edges-per-tile
#define TILES_PER_WG 16
#define WGS (N_TILES_TOTAL / TILES_PER_WG)   // 256 workgroups
#define THREADS 512                      // 16 waves (wave32)

// single-instruction ReLU (bypasses fmaxnum canonicalize lowering)
__device__ __forceinline__ float relu1(float x) {
  float y;
  asm("v_max_num_f32 %0, 0, %1" : "=v"(y) : "v"(x));
  return y;
}

__global__ __launch_bounds__(THREADS)
void ecmp_kernel(const float* __restrict__ node_j,
                 const float* __restrict__ edge_ij,
                 const float* __restrict__ W,
                 const float* __restrict__ bias,
                 float* __restrict__ out)
{
  // W as bf16, pre-swizzled into per-lane WMMA A-fragment order: 128 KB
  __shared__ __align__(32) unsigned short sWfrag[64 * 2 * 32 * 16];
  // edge tile bf16 in B-fragment order, double buffered: 2 x 2 KB
  __shared__ __align__(32) unsigned short sEfrag[2][2 * 32 * 16];
  // node tile f32 [16 edges][32], double buffered: 2 x 2 KB
  __shared__ __align__(32) float sNode[2][16 * 32];

  const int tid  = threadIdx.x;
  const int lane = tid & 31;
  const int wv   = tid >> 5;             // wave 0..15 owns output rows i = 2wv, 2wv+1
  const int lo   = lane & 15;
  const int hi   = lane >> 4;
  const int g0   = blockIdx.x * TILES_PER_WG;

  // ---- Stage W: each thread owns one even column pair (col, col+1) and walks
  // all 64 feature rows k. Column address math is loop-invariant; k-dependent
  // offsets are compile-time constants in the unrolled body.
  // Fragment layout (16-bit A 16x32, ISA 7.12.2): K = j + (j&8) + 8*lhi
  //   =>  lhi=(K>>3)&1, j=(K&7)|((K>>4)<<3).
  {
    const int colW = 2 * tid;            // even, covers 0..1022
    const int tW = colW >> 4, lloW = colW & 15;
    const float* wsrc = W + colW;
    unsigned short* wbase = &sWfrag[(tW * 2) * 32 * 16 + lloW * 16];
    #pragma unroll 8
    for (int k = 0; k < 64; ++k) {
      int s   = k >> 5, Kl = k & 31;
      int lhi = (Kl >> 3) & 1;
      int j   = (Kl & 7) | ((Kl >> 1) & 8);
      int d   = (s * 32 + lhi * 16) * 16 + j;       // compile-time per iter
      float2 v = *reinterpret_cast<const float2*>(wsrc + (long)k * NN);
      *reinterpret_cast<__bf16*>(&wbase[d])      = (__bf16)v.x;
      *reinterpret_cast<__bf16*>(&wbase[d + 16]) = (__bf16)v.y;  // col+1, same j
    }
  }

  // ---- Precompute this thread's paired edge-fragment (src,dst) offsets:
  // dest slots 2*tid, 2*tid+1 share (ln,s), adjacent j => adjacent source.
  const int eDst = tid * 2;
  int eSrc;
  {
    int j0 = eDst & 15, ln = (eDst >> 4) & 31, s = eDst >> 9;
    eSrc = (ln & 15) * EDGE_DIM + 32 * s + (j0 & 7) + 16 * (j0 >> 3) + 8 * (ln >> 4);  // even
  }

  // ---- Stage tile 0 ----
  {
    long b64 = (long)g0 * 16 * EDGE_DIM;
    float2 v = *reinterpret_cast<const float2*>(&edge_ij[b64 + eSrc]);
    v2bf pk; pk.x = (__bf16)v.x; pk.y = (__bf16)v.y;     // v_cvt_pk_bf16_f32
    *reinterpret_cast<v2bf*>(&sEfrag[0][eDst]) = pk;
    if (tid < 128)
      reinterpret_cast<float4*>(sNode[0])[tid] =
          reinterpret_cast<const float4*>(node_j + (long)g0 * 512)[tid];
  }

  // ---- Bias folded into WMMA C-initializer (tile-invariant):
  // C(M=r+8hi, N) starts at b[32i + 16p + 8hi + r], independent of N.
  v8f cini[2][2];
  #pragma unroll
  for (int ii = 0; ii < 2; ++ii)
    #pragma unroll
    for (int p = 0; p < 2; ++p) {
      int fb = 32 * (2 * wv + ii) + 16 * p + 8 * hi;
      #pragma unroll
      for (int r = 0; r < 8; ++r) cini[ii][p][r] = bias[fb + r];
    }

  __syncthreads();

  // ---- Preload this wave's 8 W fragments into registers (reused all tiles) ----
  v16bf wf[2][2][2];
  #pragma unroll
  for (int ii = 0; ii < 2; ++ii)
    #pragma unroll
    for (int p = 0; p < 2; ++p)
      #pragma unroll
      for (int s = 0; s < 2; ++s) {
        int t = 4 * wv + 2 * ii + p;
        wf[ii][p][s] =
          *reinterpret_cast<const v16bf*>(&sWfrag[((t * 2 + s) * 32 + lane) * 16]);
      }

  int buf = 0;
  for (int tt = 0; tt < TILES_PER_WG; ++tt) {
    __syncthreads();   // staged buf visible; prev buf fully consumed (1 barrier/tile)

    // ---- Stage next tile (precomputed offsets, packed cvt, b32 LDS store) ----
    if (tt + 1 < TILES_PER_WG) {
      long b64 = (long)(g0 + tt + 1) * 16 * EDGE_DIM;
      float2 v = *reinterpret_cast<const float2*>(&edge_ij[b64 + eSrc]);
      v2bf pk; pk.x = (__bf16)v.x; pk.y = (__bf16)v.y;
      *reinterpret_cast<v2bf*>(&sEfrag[buf ^ 1][eDst]) = pk;
      if (tid < 128)
        reinterpret_cast<float4*>(sNode[buf ^ 1])[tid] =
            reinterpret_cast<const float4*>(node_j + (long)(g0 + tt + 1) * 512)[tid];
    }
    if (tt + 2 < TILES_PER_WG) {         // global_prefetch_b8 two tiles ahead
      __builtin_prefetch(&edge_ij[(long)(g0 + tt + 2) * 16 * EDGE_DIM] + tid * 2, 0, 0);
    }

    // ---- B fragments + node values for this tile ----
    v16bf be0 = *reinterpret_cast<const v16bf*>(&sEfrag[buf][(0 * 32 + lane) * 16]);
    v16bf be1 = *reinterpret_cast<const v16bf*>(&sEfrag[buf][(1 * 32 + lane) * 16]);
    float nv[2][8];
    {
      const float* np = &sNode[buf][lo * 32 + 8 * hi];
      #pragma unroll
      for (int p = 0; p < 2; ++p)
        #pragma unroll
        for (int r = 0; r < 8; ++r) nv[p][r] = np[16 * p + r];
    }

    // ---- 8 WMMAs, s-major: 4 independent accumulator chains ----
    v8f c[2][2];
    #pragma unroll
    for (int ii = 0; ii < 2; ++ii)
      #pragma unroll
      for (int p = 0; p < 2; ++p) c[ii][p] = cini[ii][p];

    #pragma unroll
    for (int s = 0; s < 2; ++s) {
      v16bf be = (s == 0) ? be0 : be1;
      #pragma unroll
      for (int ii = 0; ii < 2; ++ii)
        #pragma unroll
        for (int p = 0; p < 2; ++p)
          c[ii][p] = __builtin_amdgcn_wmma_f32_16x16x32_bf16(
              false, wf[ii][p][s], false, be, (short)0, c[ii][p], false, false);
    }

    // ---- Epilogue: 1-op ReLU + dot with node; direct global store ----
    long ob = (long)(g0 + tt) * 512 + lo * 32 + 2 * wv;
    #pragma unroll
    for (int ii = 0; ii < 2; ++ii) {
      float a0 = 0.0f, a1 = 0.0f;
      #pragma unroll
      for (int r = 0; r < 8; ++r) {
        a0 = fmaf(relu1(c[ii][0][r]), nv[0][r], a0);
        a1 = fmaf(relu1(c[ii][1][r]), nv[1][r], a1);
      }
      float acc = a0 + a1;
      // reduce across lane halves (M=0..7 vs 8..15): SWAPX16 ds_swizzle
      acc += __int_as_float(__builtin_amdgcn_ds_swizzle(__float_as_int(acc), 0x401F));
      if (hi == 0) out[ob + ii] = acc;   // out[(g*16+lo)*32 + 2wv+ii]
    }
    buf ^= 1;
  }
}

extern "C" void kernel_launch(void* const* d_in, const int* in_sizes, int n_in,
                              void* d_out, int out_size, void* d_ws, size_t ws_size,
                              hipStream_t stream) {
  const float* node_j  = (const float*)d_in[0];
  const float* edge_ij = (const float*)d_in[1];
  const float* W       = (const float*)d_in[2];
  const float* b       = (const float*)d_in[3];
  float* out = (float*)d_out;
  (void)in_sizes; (void)n_in; (void)out_size; (void)d_ws; (void)ws_size;
  ecmp_kernel<<<WGS, THREADS, 0, stream>>>(node_j, edge_ij, W, b, out);
}
